// _LlamaAttention_26199300505689
// MI455X (gfx1250) — matrix operand order repaired and checked by reading, not compiled, not run on hardware
//
#include <hip/hip_runtime.h>
#include <hip/hip_bf16.h>
#include <math.h>

// ---------------------------------------------------------------------------
// Problem constants (match reference)
// ---------------------------------------------------------------------------
#define TSEQ 2048
#define HDIM 4096
#define NQH  32
#define NKVH 8
#define HD   128
#define QDIM (NQH * HD)    // 4096
#define KVDIM (NKVH * HD)  // 1024

typedef __attribute__((ext_vector_type(16))) __bf16 v16bf;
typedef __attribute__((ext_vector_type(8)))  __bf16 v8bf;
typedef __attribute__((ext_vector_type(8)))  float  v8f;
typedef int v4i_vs __attribute__((__vector_size__(4 * sizeof(int))));

#define WMMA_BF16(A, B, C) \
  __builtin_amdgcn_wmma_f32_16x16x32_bf16(false, (A), false, (B), (short)0, (C), false, false)

// gfx1250 async global->LDS copy (ASYNCcnt-tracked), if the toolchain has it.
#if defined(__has_builtin)
#if __has_builtin(__builtin_amdgcn_global_load_async_to_lds_b128)
#define HAVE_ASYNC_LDS 1
#endif
#endif

#ifdef HAVE_ASYNC_LDS
#define ASYNC_COPY_B128(gsrc, ldst)                                         \
  __builtin_amdgcn_global_load_async_to_lds_b128(                           \
      (__attribute__((address_space(1))) v4i_vs*)(gsrc),                    \
      (__attribute__((address_space(3))) v4i_vs*)(ldst), 0, 0)
#define ASYNC_WAIT() asm volatile("s_wait_asynccnt 0" ::: "memory")
#else
#define ASYNC_WAIT()
#endif

// Build a 16x32 A-fragment / 32x16 B-fragment from two 16B loads.
static __device__ __forceinline__ v16bf ld_frag(const __bf16* lo_p, const __bf16* hi_p) {
  v8bf lo = *(const v8bf*)lo_p;
  v8bf hi = *(const v8bf*)hi_p;
  v16bf r;
#pragma unroll
  for (int e = 0; e < 8; ++e) { r[e] = lo[e]; r[8 + e] = hi[e]; }
  return r;
}

// ---------------------------------------------------------------------------
// f32 -> bf16 down-convert (grid-stride)
// ---------------------------------------------------------------------------
__global__ void cvt_f32_bf16(const float* __restrict__ x, __bf16* __restrict__ y, size_t n) {
  size_t i = (size_t)blockIdx.x * blockDim.x + threadIdx.x;
  size_t stride = (size_t)gridDim.x * blockDim.x;
  for (; i < n; i += stride) y[i] = (__bf16)x[i];
}

// ---------------------------------------------------------------------------
// RoPE (rotate-half, HF convention) + convert to bf16.
// ---------------------------------------------------------------------------
__global__ void rope_cvt(const float* __restrict__ X, const int* __restrict__ pos,
                         __bf16* __restrict__ Y, int nheads) {
  int idx = blockIdx.x * blockDim.x + threadIdx.x;
  int total = TSEQ * nheads * 64;
  if (idx >= total) return;
  int i = idx & 63;
  int h = (idx >> 6) % nheads;
  int t = idx / (nheads * 64);
  float p = (float)pos[t];
  float freq = __expf(-(float)(2 * i) * (1.0f / 128.0f) * __logf(500000.0f));
  float ang = p * freq;
  float s, c;
  __sincosf(ang, &s, &c);
  size_t base = (size_t)t * nheads * 128 + (size_t)h * 128;
  float x1 = X[base + i];
  float x2 = X[base + i + 64];
  Y[base + i]      = (__bf16)(x1 * c - x2 * s);
  Y[base + i + 64] = (__bf16)(x2 * c + x1 * s);
}

// ---------------------------------------------------------------------------
// bf16 GEMM: C[M,N] f32 = A[M,K] bf16 @ B[K,N] bf16 (both row-major).
// Block 128x128, 8 waves in 2x4 grid, wave tile 64x32, K-step 32.
// Double-buffered LDS + software pipeline.
// ---------------------------------------------------------------------------
#define GM 128
#define GN 128
#define GK 32

__global__ __launch_bounds__(256) void gemm_bf16(const __bf16* __restrict__ A,
                                                 const __bf16* __restrict__ B,
                                                 float* __restrict__ C,
                                                 int M, int N, int K) {
  __shared__ __bf16 sA[2][GM][GK];    // row-major A tiles
  __shared__ __bf16 sBT[2][GN][GK];   // B tiles stored transposed [n][k]

  const int tid  = threadIdx.x;
  const int lane = tid & 31;
  const int wave = tid >> 5;
  const int l15  = lane & 15;
  const int lh   = lane >> 4;
  const int wm   = (wave & 1) * 64;
  const int wn   = (wave >> 1) * 32;
  const int bm   = blockIdx.y * GM;
  const int bn   = blockIdx.x * GN;

  const int arow = tid >> 1;          // 0..127
  const int acol = (tid & 1) * 16;    // 0 / 16
  const int brow = tid >> 4;          // 0..15 (+16 for second row)
  const int bcol = (tid & 15) * 8;    // 0..120

  v8f acc[4][2];
#pragma unroll
  for (int i = 0; i < 4; ++i)
#pragma unroll
    for (int j = 0; j < 2; ++j)
#pragma unroll
      for (int e = 0; e < 8; ++e) acc[i][j][e] = 0.0f;

  v8bf nb0, nb1;
#ifndef HAVE_ASYNC_LDS
  v8bf na0, na1;
#endif

  // ---- prologue: stage K-tile 0 into buffer 0
  {
    const __bf16* ap = A + (size_t)(bm + arow) * K + acol;
    const __bf16* bp = B + (size_t)brow * N + bn + bcol;
#ifdef HAVE_ASYNC_LDS
    ASYNC_COPY_B128(ap,     &sA[0][arow][acol]);
    ASYNC_COPY_B128(ap + 8, &sA[0][arow][acol + 8]);
#else
    na0 = *(const v8bf*)ap;
    na1 = *(const v8bf*)(ap + 8);
    *(v8bf*)&sA[0][arow][acol]     = na0;
    *(v8bf*)&sA[0][arow][acol + 8] = na1;
#endif
    nb0 = *(const v8bf*)bp;
    nb1 = *(const v8bf*)(bp + (size_t)16 * N);
#pragma unroll
    for (int e = 0; e < 8; ++e) {
      sBT[0][bcol + e][brow]      = nb0[e];
      sBT[0][bcol + e][brow + 16] = nb1[e];
    }
  }
  ASYNC_WAIT();
  __syncthreads();

  int buf = 0;
  for (int k0 = 0; k0 < K; k0 += GK, buf ^= 1) {
    const bool more = (k0 + GK) < K;
    // ---- issue next tile's global loads before computing this one
    if (more) {
      const __bf16* ap = A + (size_t)(bm + arow) * K + k0 + GK + acol;
      const __bf16* bp = B + (size_t)(k0 + GK + brow) * N + bn + bcol;
#ifdef HAVE_ASYNC_LDS
      ASYNC_COPY_B128(ap,     &sA[buf ^ 1][arow][acol]);
      ASYNC_COPY_B128(ap + 8, &sA[buf ^ 1][arow][acol + 8]);
#else
      na0 = *(const v8bf*)ap;
      na1 = *(const v8bf*)(ap + 8);
#endif
      nb0 = *(const v8bf*)bp;
      nb1 = *(const v8bf*)(bp + (size_t)16 * N);
    }

    // ---- 8 WMMAs from the current buffer
    v16bf af[4], bfr[2];
#pragma unroll
    for (int i = 0; i < 4; ++i) {
      const __bf16* ap = &sA[buf][wm + i * 16 + l15][lh * 8];
      af[i] = ld_frag(ap, ap + 16);
    }
#pragma unroll
    for (int j = 0; j < 2; ++j) {
      const __bf16* bp = &sBT[buf][wn + j * 16 + l15][lh * 8];
      bfr[j] = ld_frag(bp, bp + 16);
    }
#pragma unroll
    for (int i = 0; i < 4; ++i)
#pragma unroll
      for (int j = 0; j < 2; ++j)
        acc[i][j] = WMMA_BF16(af[i], bfr[j], acc[i][j]);

    // ---- stash next tile into the other buffer, drain, barrier
    if (more) {
#ifndef HAVE_ASYNC_LDS
      *(v8bf*)&sA[buf ^ 1][arow][acol]     = na0;
      *(v8bf*)&sA[buf ^ 1][arow][acol + 8] = na1;
#endif
#pragma unroll
      for (int e = 0; e < 8; ++e) {
        sBT[buf ^ 1][bcol + e][brow]      = nb0[e];
        sBT[buf ^ 1][bcol + e][brow + 16] = nb1[e];
      }
    }
    ASYNC_WAIT();
    __syncthreads();
  }

  // D layout: lane<16 -> n=l15, m=r ; lane>=16 -> n=l15, m=r+8
#pragma unroll
  for (int i = 0; i < 4; ++i)
#pragma unroll
    for (int j = 0; j < 2; ++j)
#pragma unroll
      for (int r = 0; r < 8; ++r) {
        int row = bm + wm + i * 16 + r + (lh << 3);
        int col = bn + wn + j * 16 + l15;
        C[(size_t)row * N + col] = acc[i][j][r];
      }
}

// ---------------------------------------------------------------------------
// Causal flash attention with GQA, double-buffered K/V tiles.
// Grid: (T/128, NQH). Block 256 = 8 waves; wave w owns 16 query rows.
// ---------------------------------------------------------------------------
__global__ __launch_bounds__(256) void flash_attn(const __bf16* __restrict__ Qb,
                                                  const __bf16* __restrict__ Kb,
                                                  const __bf16* __restrict__ Vb,
                                                  __bf16* __restrict__ Ob) {
  __shared__ __bf16 sK[2][64][128];    // K tile row-major == B^T layout for Q@K^T
  __shared__ __bf16 sVT[2][128][64];   // V tile transposed == B^T layout for P@V
  __shared__ __bf16 sP[8][16][64];     // per-wave P scratch (D-layout -> A-layout)

  const int tid  = threadIdx.x;
  const int lane = tid & 31;
  const int wave = tid >> 5;
  const int l15  = lane & 15;
  const int lh   = lane >> 4;

  const int h     = blockIdx.y;
  const int kvh   = h >> 2;            // GQA group of 4
  const int qb0   = blockIdx.x * 128;
  const int qrow0 = qb0 + wave * 16;

  const int krow = tid >> 2;           // 0..63
  const int kcb  = (tid & 3) * 32;     // 0/32/64/96

  // Q fragments: 16 rows x hd=128 as 4 K-chunks of 32
  v16bf aQ[4];
#pragma unroll
  for (int kc = 0; kc < 4; ++kc) {
    const __bf16* qp = Qb + (size_t)(qrow0 + l15) * QDIM + h * HD + kc * 32 + lh * 8;
    aQ[kc] = ld_frag(qp, qp + 16);
  }

  v8f accO[8];
#pragma unroll
  for (int t = 0; t < 8; ++t)
#pragma unroll
    for (int e = 0; e < 8; ++e) accO[t][e] = 0.0f;

  float mrun[8], lrun[8];
#pragma unroll
  for (int r = 0; r < 8; ++r) { mrun[r] = -3.0e38f; lrun[r] = 0.0f; }

  const float scale = 0.08838834764831845f;  // 1/sqrt(128)
  const int send = qb0 + 128;                // causal: tiles up to diagonal

  v8bf vr[4];
#ifndef HAVE_ASYNC_LDS
  v8bf kr[4];
#endif

  auto g_issue = [&](int s0, int dbuf) {
    const __bf16* kp = Kb + (size_t)(s0 + krow) * KVDIM + kvh * HD + kcb;
    const __bf16* vp = Vb + (size_t)(s0 + krow) * KVDIM + kvh * HD + kcb;
#ifdef HAVE_ASYNC_LDS
#pragma unroll
    for (int j = 0; j < 4; ++j)
      ASYNC_COPY_B128(kp + j * 8, &sK[dbuf][krow][kcb + j * 8]);
#else
#pragma unroll
    for (int j = 0; j < 4; ++j) kr[j] = *(const v8bf*)(kp + j * 8);
#endif
#pragma unroll
    for (int j = 0; j < 4; ++j) vr[j] = *(const v8bf*)(vp + j * 8);
  };
  auto stash = [&](int dbuf) {
#ifndef HAVE_ASYNC_LDS
#pragma unroll
    for (int j = 0; j < 4; ++j) *(v8bf*)&sK[dbuf][krow][kcb + j * 8] = kr[j];
#endif
#pragma unroll
    for (int j = 0; j < 4; ++j)
#pragma unroll
      for (int e = 0; e < 8; ++e) sVT[dbuf][kcb + j * 8 + e][krow] = vr[j][e];
  };

  // ---- prologue
  g_issue(0, 0);
  stash(0);
  ASYNC_WAIT();
  __syncthreads();

  int buf = 0;
  for (int s0 = 0; s0 < send; s0 += 64, buf ^= 1) {
    const bool more = (s0 + 64) < send;
    if (more) g_issue(s0 + 64, buf ^ 1);  // overlap with compute below

    // ---- S = Q @ K^T : 16x64 per wave
    v8f accS[4];
#pragma unroll
    for (int j = 0; j < 4; ++j)
#pragma unroll
      for (int e = 0; e < 8; ++e) accS[j][e] = 0.0f;
#pragma unroll
    for (int j = 0; j < 4; ++j)
#pragma unroll
      for (int kc = 0; kc < 4; ++kc) {
        const __bf16* bp = &sK[buf][j * 16 + l15][kc * 32 + lh * 8];
        v16bf b = ld_frag(bp, bp + 16);
        accS[j] = WMMA_BF16(aQ[kc], b, accS[j]);
      }

    // ---- masked online softmax (D layout: col=l15, row=r+8*lh)
    float sc[4][8];
    float rmax[8];
#pragma unroll
    for (int r = 0; r < 8; ++r) rmax[r] = -3.0e38f;
#pragma unroll
    for (int j = 0; j < 4; ++j)
#pragma unroll
      for (int r = 0; r < 8; ++r) {
        float v = accS[j][r] * scale;
        sc[j][r] = v;
        bool ok = (s0 + j * 16 + l15) <= (qrow0 + r + (lh << 3));
        rmax[r] = fmaxf(rmax[r], ok ? v : -3.0e38f);
      }
#pragma unroll
    for (int r = 0; r < 8; ++r)
#pragma unroll
      for (int mk = 8; mk >= 1; mk >>= 1)
        rmax[r] = fmaxf(rmax[r], __shfl_xor(rmax[r], mk, 32));

    float fsc[8], rsum[8];
#pragma unroll
    for (int r = 0; r < 8; ++r) {
      float mnew = fmaxf(mrun[r], rmax[r]);
      fsc[r]  = __expf(mrun[r] - mnew);
      mrun[r] = mnew;
      rsum[r] = 0.0f;
    }
#pragma unroll
    for (int j = 0; j < 4; ++j)
#pragma unroll
      for (int r = 0; r < 8; ++r) {
        bool ok = (s0 + j * 16 + l15) <= (qrow0 + r + (lh << 3));
        float pv = ok ? __expf(sc[j][r] - mrun[r]) : 0.0f;
        sc[j][r] = pv;
        rsum[r] += pv;
      }
#pragma unroll
    for (int r = 0; r < 8; ++r) {
#pragma unroll
      for (int mk = 8; mk >= 1; mk >>= 1)
        rsum[r] += __shfl_xor(rsum[r], mk, 32);
      lrun[r] = lrun[r] * fsc[r] + rsum[r];
    }
#pragma unroll
    for (int t = 0; t < 8; ++t)
#pragma unroll
      for (int r = 0; r < 8; ++r) accO[t][r] *= fsc[r];

    // ---- P: D layout -> A layout via wave-private LDS
#pragma unroll
    for (int j = 0; j < 4; ++j)
#pragma unroll
      for (int r = 0; r < 8; ++r)
        sP[wave][r + (lh << 3)][j * 16 + l15] = (__bf16)sc[j][r];
    asm volatile("s_wait_dscnt 0" ::: "memory");  // wave-local LDS RAW fence

    // ---- O += P @ V
#pragma unroll
    for (int tn = 0; tn < 8; ++tn)
#pragma unroll
      for (int kc = 0; kc < 2; ++kc) {
        const __bf16* pp = &sP[wave][l15][kc * 32 + lh * 8];
        v16bf aP = ld_frag(pp, pp + 16);
        const __bf16* vb = &sVT[buf][tn * 16 + l15][kc * 32 + lh * 8];
        v16bf bV = ld_frag(vb, vb + 16);
        accO[tn] = WMMA_BF16(aP, bV, accO[tn]);
      }

    if (more) stash(buf ^ 1);
    ASYNC_WAIT();
    __syncthreads();
  }

  // ---- normalize and store (bf16)
#pragma unroll
  for (int tn = 0; tn < 8; ++tn)
#pragma unroll
    for (int r = 0; r < 8; ++r) {
      int row = qrow0 + r + (lh << 3);
      int col = h * HD + tn * 16 + l15;
      Ob[(size_t)row * QDIM + col] = (__bf16)(accO[tn][r] / lrun[r]);
    }
}

// ---------------------------------------------------------------------------
// Host orchestration
// ---------------------------------------------------------------------------
extern "C" void kernel_launch(void* const* d_in, const int* in_sizes, int n_in,
                              void* d_out, int out_size, void* d_ws, size_t ws_size,
                              hipStream_t stream) {
  (void)in_sizes; (void)n_in; (void)out_size; (void)ws_size;
  const float* hs  = (const float*)d_in[0];
  const int*   pos = (const int*)d_in[1];
  const float* wq  = (const float*)d_in[2];
  const float* wk  = (const float*)d_in[3];
  const float* wv  = (const float*)d_in[4];
  const float* wo  = (const float*)d_in[5];
  float* out = (float*)d_out;

  char* p = (char*)d_ws;
  auto take = [&](size_t n) { char* r = p; p += (n + 255) & ~(size_t)255; return r; };
  __bf16* Xb  = (__bf16*)take((size_t)TSEQ * HDIM * 2);
  __bf16* Wqb = (__bf16*)take((size_t)HDIM * QDIM * 2);
  __bf16* Wkb = (__bf16*)take((size_t)HDIM * KVDIM * 2);
  __bf16* Wvb = (__bf16*)take((size_t)HDIM * KVDIM * 2);
  __bf16* Wob = (__bf16*)take((size_t)QDIM * HDIM * 2);
  float*  Qf  = (float*) take((size_t)TSEQ * QDIM * 4);
  float*  Kf  = (float*) take((size_t)TSEQ * KVDIM * 4);
  float*  Vf  = (float*) take((size_t)TSEQ * KVDIM * 4);
  __bf16* Qb  = (__bf16*)take((size_t)TSEQ * QDIM * 2);
  __bf16* Kb  = (__bf16*)take((size_t)TSEQ * KVDIM * 2);
  __bf16* Vb  = (__bf16*)take((size_t)TSEQ * KVDIM * 2);
  __bf16* AOb = (__bf16*)take((size_t)TSEQ * QDIM * 2);

  cvt_f32_bf16<<<2048, 256, 0, stream>>>(hs, Xb, (size_t)TSEQ * HDIM);
  cvt_f32_bf16<<<2048, 256, 0, stream>>>(wq, Wqb, (size_t)HDIM * QDIM);
  cvt_f32_bf16<<<2048, 256, 0, stream>>>(wk, Wkb, (size_t)HDIM * KVDIM);
  cvt_f32_bf16<<<2048, 256, 0, stream>>>(wv, Wvb, (size_t)HDIM * KVDIM);
  cvt_f32_bf16<<<2048, 256, 0, stream>>>(wo, Wob, (size_t)QDIM * HDIM);

  gemm_bf16<<<dim3(QDIM / GN, TSEQ / GM), 256, 0, stream>>>(Xb, Wqb, Qf, TSEQ, QDIM, HDIM);
  gemm_bf16<<<dim3(KVDIM / GN, TSEQ / GM), 256, 0, stream>>>(Xb, Wkb, Kf, TSEQ, KVDIM, HDIM);
  gemm_bf16<<<dim3(KVDIM / GN, TSEQ / GM), 256, 0, stream>>>(Xb, Wvb, Vf, TSEQ, KVDIM, HDIM);

  rope_cvt<<<(TSEQ * NQH * 64) / 256, 256, 0, stream>>>(Qf, pos, Qb, NQH);
  rope_cvt<<<(TSEQ * NKVH * 64) / 256, 256, 0, stream>>>(Kf, pos, Kb, NKVH);
  cvt_f32_bf16<<<2048, 256, 0, stream>>>(Vf, Vb, (size_t)TSEQ * KVDIM);

  flash_attn<<<dim3(TSEQ / 128, NQH), 256, 0, stream>>>(Qb, Kb, Vb, AOb);

  gemm_bf16<<<dim3(HDIM / GN, TSEQ / GM), 256, 0, stream>>>(AOb, Wob, out, TSEQ, HDIM, QDIM);
}
